// transformer_11940009083280
// MI455X (gfx1250) — matrix operand order repaired and checked by reading, not compiled, not run on hardware
//
#include <hip/hip_runtime.h>
#include <hip/hip_bf16.h>
#include <float.h>

// Problem constants (fixed by the reference)
#define BB   4
#define SEQ  1024
#define DIMC 1024
#define NH   16
#define HD   64
#define NL   3
#define FFC  4096
#define ROWS (BB * SEQ)          // 4096 token rows

typedef __attribute__((ext_vector_type(16))) __bf16 v16bf;
typedef __attribute__((ext_vector_type(8)))  float  v8f;

union FragU { v16bf v; uint4 q[2]; };

__device__ __forceinline__ unsigned short f2bf(float f) {
  unsigned int u = __float_as_uint(f);
  u += 0x7fffu + ((u >> 16) & 1u);          // round-to-nearest-even
  return (unsigned short)(u >> 16);
}

// ---------------------------------------------------------------------------
// Utility kernels
// ---------------------------------------------------------------------------
__global__ __launch_bounds__(256) void cvt_f32_bf16(const float* __restrict__ s,
                                                    unsigned short* __restrict__ d,
                                                    long long n) {
  long long i = (long long)blockIdx.x * 256 + threadIdx.x;
  if (i < n) d[i] = f2bf(s[i]);
}

__global__ __launch_bounds__(256) void copy_f32(const float* __restrict__ s,
                                                float* __restrict__ d,
                                                long long n) {
  long long i = (long long)blockIdx.x * 256 + threadIdx.x;
  if (i < n) d[i] = s[i];
}

// ---------------------------------------------------------------------------
// LayerNorm: one block per token row, fp32 in -> bf16 out
// ---------------------------------------------------------------------------
__global__ __launch_bounds__(256) void layernorm_bf16(const float* __restrict__ x,
                                                      const float* __restrict__ g,
                                                      const float* __restrict__ b,
                                                      unsigned short* __restrict__ out) {
  __shared__ float red[256];
  const int row = blockIdx.x;
  const int t = threadIdx.x;
  const float* xr = x + (size_t)row * DIMC;
  float v0 = xr[t], v1 = xr[t + 256], v2 = xr[t + 512], v3 = xr[t + 768];

  red[t] = v0 + v1 + v2 + v3;
  __syncthreads();
  for (int o = 128; o > 0; o >>= 1) { if (t < o) red[t] += red[t + o]; __syncthreads(); }
  const float mu = red[0] * (1.0f / DIMC);
  __syncthreads();

  float d0 = v0 - mu, d1 = v1 - mu, d2 = v2 - mu, d3 = v3 - mu;
  red[t] = d0 * d0 + d1 * d1 + d2 * d2 + d3 * d3;
  __syncthreads();
  for (int o = 128; o > 0; o >>= 1) { if (t < o) red[t] += red[t + o]; __syncthreads(); }
  const float inv = rsqrtf(red[0] * (1.0f / DIMC) + 1e-5f);

  unsigned short* orow = out + (size_t)row * DIMC;
  orow[t]       = f2bf(d0 * inv * g[t]       + b[t]);
  orow[t + 256] = f2bf(d1 * inv * g[t + 256] + b[t + 256]);
  orow[t + 512] = f2bf(d2 * inv * g[t + 512] + b[t + 512]);
  orow[t + 768] = f2bf(d3 * inv * g[t + 768] + b[t + 768]);
}

// ---------------------------------------------------------------------------
// WMMA bf16 GEMM: C = alpha * A(MxK) @ B(KxN) [+ bias][+ resid]
// Block tile 128 x BN x 32, 8 waves (4 along M, 2 along N), wave tile
// 32 x (BN/2) -> acc[2][BN/32] of v_wmma_f32_16x16x32_bf16.
// Software-pipelined: global loads for step k+1 issue before the WMMAs of
// step k (register staging), LDS stores after the read barrier.
//   AF32: A operand is f32, converted to bf16 while staging (attn @ V)
//   BT:   B source is [N][K] row-major (K-matrix), else [K][N] (transposed
//         while staging into Bs[n][k])
// ---------------------------------------------------------------------------
#define OUT_F32      0   // f32 store (batched)
#define OUT_F32_BIAS 1   // f32 store + bias[col]
#define OUT_RESID    2   // f32 store + resid[row,col] (+bias) (in-place safe)
#define OUT_QKV      3   // scatter bf16 into q/k/v [b,h,n,d]
#define OUT_BHND     4   // bf16 store at [b, row, h*HD + col], z = b*NH + h

template <int BN, int AF32, int BT>
__global__ __launch_bounds__(256) void gemm_wmma_bf16(
    const void* __restrict__ Aptr, long long sAz, int lda,
    const unsigned short* __restrict__ Bptr, long long sBz, int ldb,
    const float* __restrict__ bias, const float* resid,
    void* Out, int mode, long long sCz, int ldc,
    int M, int Nn, int K, float alpha,
    unsigned short* __restrict__ q16, unsigned short* __restrict__ k16,
    unsigned short* __restrict__ v16o) {
  constexpr int NT  = BN / 32;               // 16-wide col tiles per wave (4 | 2)
  constexpr int BCH = (BN * 32) / (256 * 8); // B staging chunks per thread (2 | 1)
  // Padded rows: 40 * 2B = 80B stride keeps every 16B fragment chunk aligned.
  __shared__ __align__(16) unsigned short As[128][40];
  __shared__ __align__(16) unsigned short Bs[BN][40];

  const int tid  = threadIdx.x;
  const int lane = tid & 31;
  const int w    = tid >> 5;     // 8 waves
  const int wm   = w & 3;        // 4 along M
  const int wn   = w >> 2;       // 2 along N
  const int half = lane >> 4;
  const int l16  = lane & 15;

  const int row0 = blockIdx.y * 128;
  const int col0 = blockIdx.x * BN;
  const int z    = blockIdx.z;

  v8f acc[2][NT] = {};

  const unsigned short* Abf = (const unsigned short*)Aptr + (AF32 ? 0 : (size_t)z * sAz);
  const float*          Af  = (const float*)Aptr + (size_t)z * sAz;
  const unsigned short* Bb  = Bptr + (size_t)z * sBz;

  // register staging for the software pipeline
  uint4  aR[2];
  float4 aRf[2][2];
  uint4  bR[BCH];

  auto loadRegs = [&](int k0) {
#pragma unroll
    for (int it = 0; it < 2; ++it) {
      const int c = tid + it * 256;
      const int r = c >> 2, ko = (c & 3) << 3;
      if (!AF32) {
        aR[it] = *(const uint4*)(Abf + (size_t)(row0 + r) * lda + (k0 + ko));
      } else {
        const float* s = Af + (size_t)(row0 + r) * lda + (k0 + ko);
        aRf[it][0] = *(const float4*)s;
        aRf[it][1] = *(const float4*)(s + 4);
      }
    }
#pragma unroll
    for (int it = 0; it < BCH; ++it) {
      const int c = tid + it * 256;
      if (BT) {
        const int n = c >> 2, ko = (c & 3) << 3;
        bR[it] = *(const uint4*)(Bb + (size_t)(col0 + n) * ldb + (k0 + ko));
      } else {
        const int k = c / (BN / 8), n = (c % (BN / 8)) << 3;
        bR[it] = *(const uint4*)(Bb + (size_t)(k0 + k) * ldb + (col0 + n));
      }
    }
  };

  auto storeLDS = [&]() {
#pragma unroll
    for (int it = 0; it < 2; ++it) {
      const int c = tid + it * 256;
      const int r = c >> 2, ko = (c & 3) << 3;
      if (!AF32) {
        *(uint4*)&As[r][ko] = aR[it];
      } else {
        uint4 pk;
        pk.x = (unsigned)f2bf(aRf[it][0].x) | ((unsigned)f2bf(aRf[it][0].y) << 16);
        pk.y = (unsigned)f2bf(aRf[it][0].z) | ((unsigned)f2bf(aRf[it][0].w) << 16);
        pk.z = (unsigned)f2bf(aRf[it][1].x) | ((unsigned)f2bf(aRf[it][1].y) << 16);
        pk.w = (unsigned)f2bf(aRf[it][1].z) | ((unsigned)f2bf(aRf[it][1].w) << 16);
        *(uint4*)&As[r][ko] = pk;
      }
    }
#pragma unroll
    for (int it = 0; it < BCH; ++it) {
      const int c = tid + it * 256;
      if (BT) {
        const int n = c >> 2, ko = (c & 3) << 3;
        *(uint4*)&Bs[n][ko] = bR[it];
      } else {
        const int k = c / (BN / 8), n = (c % (BN / 8)) << 3;
        unsigned short e[8];
        *(uint4*)e = bR[it];
#pragma unroll
        for (int i = 0; i < 8; ++i) Bs[n + i][k] = e[i];
      }
    }
  };

  loadRegs(0);
  storeLDS();

  for (int k0 = 0; k0 < K; k0 += 32) {
    __syncthreads();                       // LDS tile k0 visible to all waves
    const bool more = (k0 + 32 < K);
    if (more) loadRegs(k0 + 32);           // overlap global loads with WMMAs
    if (k0 + 64 < K) {                     // L2/HBM prefetch two steps ahead
      const char* p = AF32
          ? (const char*)(Af + (size_t)(row0 + (tid >> 1)) * lda + (k0 + 64))
          : (const char*)(Abf + (size_t)(row0 + (tid >> 1)) * lda + (k0 + 64));
      __builtin_prefetch(p, 0, 1);
    }

    // ---- fragments per the CDNA5 16-bit lane layouts ----
    FragU a[2], bfr[NT];
#pragma unroll
    for (int ti = 0; ti < 2; ++ti) {
      const int m = wm * 32 + ti * 16 + l16;
      a[ti].q[0] = *(const uint4*)&As[m][8 * half];        // K = 8*half + 0..7
      a[ti].q[1] = *(const uint4*)&As[m][16 + 8 * half];   // K = 16 + 8*half + 0..7
    }
#pragma unroll
    for (int tj = 0; tj < NT; ++tj) {
      const int n = wn * (BN / 2) + tj * 16 + l16;
      bfr[tj].q[0] = *(const uint4*)&Bs[n][8 * half];      // K = 8*half + 0..7
      bfr[tj].q[1] = *(const uint4*)&Bs[n][16 + 8 * half]; // K = 16 + 8*half + 0..7
    }
#pragma unroll
    for (int ti = 0; ti < 2; ++ti)
#pragma unroll
      for (int tj = 0; tj < NT; ++tj)
        acc[ti][tj] = __builtin_amdgcn_wmma_f32_16x16x32_bf16(
            false, a[ti].v, false, bfr[tj].v, (short)0, acc[ti][tj], false, false);

    __syncthreads();                       // all waves done reading LDS tile k0
    if (more) storeLDS();
  }

  // ---- epilogue: lane holds rows (r + 8*half), col l16 of each 16x16 tile ----
#pragma unroll
  for (int ti = 0; ti < 2; ++ti) {
#pragma unroll
    for (int tj = 0; tj < NT; ++tj) {
#pragma unroll
      for (int r = 0; r < 8; ++r) {
        const int grow = row0 + wm * 32 + ti * 16 + half * 8 + r;
        const int gcol = col0 + wn * (BN / 2) + tj * 16 + l16;
        const float vv = acc[ti][tj][r] * alpha;
        if (mode == OUT_F32) {
          ((float*)Out)[(size_t)z * sCz + (size_t)grow * ldc + gcol] = vv;
        } else if (mode == OUT_F32_BIAS) {
          ((float*)Out)[(size_t)z * sCz + (size_t)grow * ldc + gcol] = vv + bias[gcol];
        } else if (mode == OUT_RESID) {
          const size_t o = (size_t)grow * ldc + gcol;
          ((float*)Out)[o] = resid[o] + vv + (bias ? bias[gcol] : 0.0f);
        } else if (mode == OUT_QKV) {
          const int bI = grow >> 10, n = grow & 1023;              // ROWS = B*SEQ
          const int h = gcol / (3 * HD), s = (gcol / HD) % 3, d = gcol & (HD - 1);
          const size_t o = ((size_t)(bI * NH + h) * SEQ + n) * HD + d;
          const unsigned short val = f2bf(vv);
          if (s == 0) q16[o] = val; else if (s == 1) k16[o] = val; else v16o[o] = val;
        } else { // OUT_BHND
          const int bI = z >> 4, h = z & 15;
          ((unsigned short*)Out)[((size_t)(bI * SEQ + grow)) * (NH * HD) + h * HD + gcol] =
              f2bf(vv);
        }
      }
    }
  }
}

// ---------------------------------------------------------------------------
// Talking-heads mix (in place): each thread owns one (b,i,j) across all heads
// pre=1: + pos_bias, then causal mask (j>i -> -FLT_MAX)
// ---------------------------------------------------------------------------
__global__ __launch_bounds__(256) void talking_heads(float* __restrict__ dots,
                                                     const float* __restrict__ w,
                                                     const float* __restrict__ bvec,
                                                     const float* __restrict__ pos_bias,
                                                     int pre) {
  __shared__ float wsm[NH * NH];
  __shared__ float bsm[NH];
  if (threadIdx.x < NH * NH) wsm[threadIdx.x] = w[threadIdx.x];
  if (threadIdx.x < NH) bsm[threadIdx.x] = bvec[threadIdx.x];
  __syncthreads();

  const size_t idx = (size_t)blockIdx.x * 256 + threadIdx.x;  // B*SEQ*SEQ
  const int b = (int)(idx >> 20);
  const size_t ij = idx & ((1u << 20) - 1);
  const int i = (int)(ij >> 10), j = (int)(ij & 1023);

  const size_t base = (size_t)b * NH * SEQ * SEQ + ij;
  float vals[NH], outv[NH];
#pragma unroll
  for (int h = 0; h < NH; ++h) vals[h] = dots[base + (size_t)h * SEQ * SEQ];
#pragma unroll
  for (int g = 0; g < NH; ++g) {
    float s = bsm[g];
#pragma unroll
    for (int h = 0; h < NH; ++h) s += vals[h] * wsm[g * NH + h];
    if (pre) {
      s += pos_bias[(size_t)g * SEQ * SEQ + ij];
      if (j > i) s = -FLT_MAX;
    }
    outv[g] = s;
  }
#pragma unroll
  for (int g = 0; g < NH; ++g) dots[base + (size_t)g * SEQ * SEQ] = outv[g];
}

// ---------------------------------------------------------------------------
// Row softmax over N=1024, one block per (b,h,i) row, in place
// ---------------------------------------------------------------------------
__global__ __launch_bounds__(256) void softmax_rows(float* __restrict__ dots) {
  __shared__ float red[256];
  float* p = dots + (size_t)blockIdx.x * SEQ;
  const int t = threadIdx.x;
  float v0 = p[t], v1 = p[t + 256], v2 = p[t + 512], v3 = p[t + 768];

  red[t] = fmaxf(fmaxf(v0, v1), fmaxf(v2, v3));
  __syncthreads();
  for (int o = 128; o > 0; o >>= 1) { if (t < o) red[t] = fmaxf(red[t], red[t + o]); __syncthreads(); }
  const float m = red[0];
  __syncthreads();

  v0 = __expf(v0 - m); v1 = __expf(v1 - m); v2 = __expf(v2 - m); v3 = __expf(v3 - m);
  red[t] = v0 + v1 + v2 + v3;
  __syncthreads();
  for (int o = 128; o > 0; o >>= 1) { if (t < o) red[t] += red[t + o]; __syncthreads(); }
  const float inv = 1.0f / red[0];
  p[t] = v0 * inv; p[t + 256] = v1 * inv; p[t + 512] = v2 * inv; p[t + 768] = v3 * inv;
}

// ---------------------------------------------------------------------------
// GLU: u = a * silu(g), p is [ROWS][2*FF] f32 -> u bf16 [ROWS][FF]
// ---------------------------------------------------------------------------
__global__ __launch_bounds__(256) void glu_silu(const float* __restrict__ p,
                                                unsigned short* __restrict__ u) {
  const size_t idx = (size_t)blockIdx.x * 256 + threadIdx.x;  // ROWS*FF
  const size_t r = idx >> 12;          // / FF
  const size_t f = idx & (FFC - 1);
  const float a = p[r * (2 * FFC) + f];
  const float g = p[r * (2 * FFC) + FFC + f];
  const float sig = 1.0f / (1.0f + __expf(-g));
  u[idx] = f2bf(a * g * sig);
}

// ---------------------------------------------------------------------------
// Host launcher
// ---------------------------------------------------------------------------
extern "C" void kernel_launch(void* const* d_in, const int* in_sizes, int n_in,
                              void* d_out, int out_size, void* d_ws, size_t ws_size,
                              hipStream_t stream) {
  (void)in_sizes; (void)n_in; (void)out_size; (void)ws_size;
  const float* x_in     = (const float*)d_in[0];
  const float* pos_bias = (const float*)d_in[1];
  // d_in[2] is the causal mask; we recompute it from indices (triu, k=1)
  const float* ln1_g = (const float*)d_in[3];
  const float* ln1_b = (const float*)d_in[4];
  const float* Wqkv  = (const float*)d_in[5];
  const float* thpw  = (const float*)d_in[6];
  const float* thpb  = (const float*)d_in[7];
  const float* thqw  = (const float*)d_in[8];
  const float* thqb  = (const float*)d_in[9];
  const float* Wout  = (const float*)d_in[10];
  const float* ln2_g = (const float*)d_in[11];
  const float* ln2_b = (const float*)d_in[12];
  const float* Wglu  = (const float*)d_in[13];
  const float* bglu  = (const float*)d_in[14];
  const float* Wff2  = (const float*)d_in[15];
  const float* bff2  = (const float*)d_in[16];

  // ---- workspace carve-up ----
  char* base = (char*)d_ws;
  size_t off = 0;
  auto take = [&](size_t bytes) { char* p = base + off; off += (bytes + 255) & ~(size_t)255; return p; };
  float*          ws_x  = (float*)take((size_t)ROWS * DIMC * 4);
  unsigned short* h16   = (unsigned short*)take((size_t)ROWS * DIMC * 2);
  unsigned short* q16   = (unsigned short*)take((size_t)BB * NH * SEQ * HD * 2);
  unsigned short* k16   = (unsigned short*)take((size_t)BB * NH * SEQ * HD * 2);
  unsigned short* v16   = (unsigned short*)take((size_t)BB * NH * SEQ * HD * 2);
  unsigned short* ao16  = (unsigned short*)take((size_t)ROWS * NH * HD * 2);
  float*          big   = (float*)take((size_t)BB * NH * SEQ * SEQ * 4);  // scores / GLU pre-act
  unsigned short* u16   = (unsigned short*)take((size_t)ROWS * FFC * 2);
  unsigned short* wqkv16 = (unsigned short*)take((size_t)NL * DIMC * 3 * NH * HD * 2);
  unsigned short* wout16 = (unsigned short*)take((size_t)NL * NH * HD * DIMC * 2);
  unsigned short* wglu16 = (unsigned short*)take((size_t)NL * DIMC * 2 * FFC * 2);
  unsigned short* wff216 = (unsigned short*)take((size_t)NL * FFC * DIMC * 2);

  auto blks = [](long long n) { return (unsigned)((n + 255) / 256); };

  // ---- weight conversion f32 -> bf16 ----
  {
    long long n;
    n = (long long)NL * DIMC * 3 * NH * HD; cvt_f32_bf16<<<blks(n), 256, 0, stream>>>(Wqkv, wqkv16, n);
    n = (long long)NL * NH * HD * DIMC;     cvt_f32_bf16<<<blks(n), 256, 0, stream>>>(Wout, wout16, n);
    n = (long long)NL * DIMC * 2 * FFC;     cvt_f32_bf16<<<blks(n), 256, 0, stream>>>(Wglu, wglu16, n);
    n = (long long)NL * FFC * DIMC;         cvt_f32_bf16<<<blks(n), 256, 0, stream>>>(Wff2, wff216, n);
  }
  // running residual x
  copy_f32<<<blks((long long)ROWS * DIMC), 256, 0, stream>>>(x_in, ws_x, (long long)ROWS * DIMC);

  for (int l = 0; l < NL; ++l) {
    // --- attention ---
    layernorm_bf16<<<ROWS, 256, 0, stream>>>(ws_x, ln1_g + l * DIMC, ln1_b + l * DIMC, h16);

    // QKV: [4096 x 1024] @ [1024 x 3072] -> scatter q/k/v bf16 [b,h,n,d]
    gemm_wmma_bf16<128, 0, 0><<<dim3(3 * NH * HD / 128, ROWS / 128, 1), 256, 0, stream>>>(
        h16, 0, DIMC, wqkv16 + (size_t)l * DIMC * 3 * NH * HD, 0, 3 * NH * HD,
        nullptr, nullptr, nullptr, OUT_QKV, 0, 0,
        ROWS, 3 * NH * HD, DIMC, 1.0f, q16, k16, v16);

    // dots = q @ k^T * D^-0.5 : batched over b*h (K pre-transposed layout)
    gemm_wmma_bf16<128, 0, 1><<<dim3(SEQ / 128, SEQ / 128, BB * NH), 256, 0, stream>>>(
        q16, (long long)SEQ * HD, HD, k16, (long long)SEQ * HD, HD,
        nullptr, nullptr, big, OUT_F32, (long long)SEQ * SEQ, SEQ,
        SEQ, SEQ, HD, 0.125f, nullptr, nullptr, nullptr);

    // talking-heads pre + pos_bias + causal mask
    talking_heads<<<blks((long long)BB * SEQ * SEQ), 256, 0, stream>>>(
        big, thpw + l * NH * NH, thpb + l * NH, pos_bias, 1);

    softmax_rows<<<BB * NH * SEQ, 256, 0, stream>>>(big);

    // talking-heads post
    talking_heads<<<blks((long long)BB * SEQ * SEQ), 256, 0, stream>>>(
        big, thqw + l * NH * NH, thqb + l * NH, pos_bias, 0);

    // out = attn @ v : batched, A is f32 (converted to bf16 while staging)
    gemm_wmma_bf16<64, 1, 0><<<dim3(HD / 64, SEQ / 128, BB * NH), 256, 0, stream>>>(
        big, (long long)SEQ * SEQ, SEQ, v16, (long long)SEQ * HD, HD,
        nullptr, nullptr, ao16, OUT_BHND, 0, 0,
        SEQ, HD, SEQ, 1.0f, nullptr, nullptr, nullptr);

    // x += attn_out @ Wout
    gemm_wmma_bf16<128, 0, 0><<<dim3(DIMC / 128, ROWS / 128, 1), 256, 0, stream>>>(
        ao16, 0, NH * HD, wout16 + (size_t)l * NH * HD * DIMC, 0, DIMC,
        nullptr, ws_x, ws_x, OUT_RESID, 0, DIMC,
        ROWS, DIMC, NH * HD, 1.0f, nullptr, nullptr, nullptr);

    // --- GLU feed-forward ---
    layernorm_bf16<<<ROWS, 256, 0, stream>>>(ws_x, ln2_g + l * DIMC, ln2_b + l * DIMC, h16);

    // p = h2 @ Wglu + bglu : [4096 x 1024] @ [1024 x 8192] (reuses `big` as f32 scratch)
    gemm_wmma_bf16<128, 0, 0><<<dim3(2 * FFC / 128, ROWS / 128, 1), 256, 0, stream>>>(
        h16, 0, DIMC, wglu16 + (size_t)l * DIMC * 2 * FFC, 0, 2 * FFC,
        bglu + l * 2 * FFC, nullptr, big, OUT_F32_BIAS, 0, 2 * FFC,
        ROWS, 2 * FFC, DIMC, 1.0f, nullptr, nullptr, nullptr);

    glu_silu<<<blks((long long)ROWS * FFC), 256, 0, stream>>>(big, u16);

    // x += u @ Wff2 + bff2
    gemm_wmma_bf16<128, 0, 0><<<dim3(DIMC / 128, ROWS / 128, 1), 256, 0, stream>>>(
        u16, 0, FFC, wff216 + (size_t)l * FFC * DIMC, 0, DIMC,
        bff2 + l * DIMC, ws_x, ws_x, OUT_RESID, 0, DIMC,
        ROWS, DIMC, FFC, 1.0f, nullptr, nullptr, nullptr);
  }

  copy_f32<<<blks((long long)ROWS * DIMC), 256, 0, stream>>>(ws_x, (float*)d_out,
                                                             (long long)ROWS * DIMC);
}
